// Modelz_41970420417799
// MI455X (gfx1250) — matrix operand order repaired and checked by reading, not compiled, not run on hardware
//
#include <hip/hip_runtime.h>
#include <hip/hip_bf16.h>
#include <math.h>

// ---------------- problem constants ----------------
#define HID    300         // H
#define HP     320         // H padded to a multiple of 32 (K-chunks)
#define LDSROW 328         // LDS row stride for h (bank skew; 328*2B = 41*16B)
#define GROW   1216        // LDS row stride for gates (4H=1200 padded, bank skew)
#define BATCH  64          // B
#define TSTEP  256         // T
#define G4H    1200        // 4*H
#define NTILE  5           // N-tiles per wave in phase 1
#define NGRP   15          // 75 N-tiles / NTILE

typedef __bf16 bf16_t;
typedef __bf16        v16bf __attribute__((ext_vector_type(16)));
typedef float         v8f   __attribute__((ext_vector_type(8)));
typedef unsigned int  u32x4 __attribute__((ext_vector_type(4)));

union Frag32B { v16bf v; u32x4 q[2]; };

static __device__ __forceinline__ float sigf(float x) {
    return 1.f / (1.f + __expf(-x));
}

// ---------------------------------------------------------------------------
// Kernel 0: convert W_ih / W_hh (f32, [1200][300]) to bf16 padded [1200][320]
// ---------------------------------------------------------------------------
__global__ __launch_bounds__(256) void k_convert_w(
    const float* __restrict__ Wih, const float* __restrict__ Whh,
    bf16_t* __restrict__ WihP, bf16_t* __restrict__ WhhP)
{
    int idx = blockIdx.x * 256 + threadIdx.x;
    if (idx >= G4H * HP) return;
    int n = idx / HP;
    int k = idx - n * HP;
    float a = (k < HID) ? Wih[n * HID + k] : 0.f;
    float b = (k < HID) ? Whh[n * HID + k] : 0.f;
    WihP[idx] = (bf16_t)a;
    WhhP[idx] = (bf16_t)b;
}

// ---------------------------------------------------------------------------
// Kernel 1: Xg[T*B, 4H] = embed[tokens] @ W_ih^T + (b_ih + b_hh)
// One wave per (M-tile, group of NTILE N-tiles); K looped in chunks of 32.
// A fragment (gathered embedding rows) is reused across NTILE B tiles.
// ---------------------------------------------------------------------------
__global__ __launch_bounds__(256) void k_gates_x(
    const int*   __restrict__ tokens,
    const float* __restrict__ embed,
    const bf16_t* __restrict__ WihP,
    const float* __restrict__ b_ih,
    const float* __restrict__ b_hh,
    float* __restrict__ Xg)
{
    const int wid  = (blockIdx.x << 3) + (threadIdx.x >> 5);   // 15360 waves total
    const int lane = threadIdx.x & 31;
    const int mt   = wid / NGRP;          // 0..1023
    const int ng   = wid - mt * NGRP;     // 0..14
    const int r16  = lane & 15;
    const int hi   = lane >> 4;

    // A-matrix row for this lane: global row m = mt*16 + (lane%16)
    const int mrow = mt * 16 + r16;
    const int tok  = tokens[mrow];
    const float* arow = embed + (size_t)tok * HID;

    v8f acc[NTILE] = {};

    for (int kc = 0; kc < 10; ++kc) {
        const int kbase = kc * 32;
        // Build A fragment in contiguous k order (same as B):
        //   element i -> K = kbase + hi*16 + i
        Frag32B af;
        #pragma unroll
        for (int i = 0; i < 16; ++i) {
            int k = kbase + (hi << 4) + i;
            float v = (k < HID) ? arow[k] : 0.f;
            af.v[i] = (bf16_t)v;
        }
        #pragma unroll
        for (int j = 0; j < NTILE; ++j) {
            // B fragment: column n, 16 contiguous bf16 (32 bytes)
            const int n = (ng * NTILE + j) * 16 + r16;
            const u32x4* bp = (const u32x4*)(WihP + (size_t)n * HP + kbase + (hi << 4));
            Frag32B bfz;
            bfz.q[0] = bp[0];
            bfz.q[1] = bp[1];
            acc[j] = __builtin_amdgcn_wmma_f32_16x16x32_bf16(
                false, af.v, false, bfz.v, (short)0, acc[j], false, false);
        }
    }

    // Store D (+ fused bias): row m = mt*16 + hi*8 + r, col n
    const int mbase = mt * 16 + (hi << 3);
    #pragma unroll
    for (int j = 0; j < NTILE; ++j) {
        const int n = (ng * NTILE + j) * 16 + r16;
        const float bias = b_ih[n] + b_hh[n];
        #pragma unroll
        for (int r = 0; r < 8; ++r)
            Xg[(size_t)(mbase + r) * G4H + n] = acc[j][r] + bias;
    }
}

// ---------------------------------------------------------------------------
// Kernel 2: serial LSTM recurrence. One persistent 1024-thread workgroup
// (32 wave32s on one WGP). Everything on the per-step critical path stays
// inside the WGP: h (bf16), c (f32) and the gate pre-activations (bf16) all
// live in LDS (268 KB of the 320 KB WGP LDS); W_hh streams from its
// L2-resident bf16 image; next step's Xg rows are prefetched during WMMA.
// Per step:
//   stage 1: gates_lds = h @ W_hh^T   (WMMA bf16, f32 acc)  + Xg[t+1] prefetch
//   barrier
//   stage 2: elementwise cell update (sigmoid/tanh), h,c refresh in LDS
//   barrier
// ---------------------------------------------------------------------------
__global__ __launch_bounds__(1024) void k_recurrence(
    const float*  __restrict__ Xg,
    const bf16_t* __restrict__ WhhP,
    const float*  __restrict__ W_lin,
    const float*  __restrict__ b_lin,
    float*        __restrict__ out)
{
    extern __shared__ char smem[];
    bf16_t* h_lds = (bf16_t*)smem;                            // [B][LDSROW] bf16
    float*  c_lds = (float*)(smem + BATCH * LDSROW * 2);      // [B][H]      f32
    bf16_t* g_lds = (bf16_t*)(smem + BATCH * LDSROW * 2
                                   + BATCH * HID * 4);        // [B][GROW]   bf16

    const int tid  = threadIdx.x;
    const int wid  = tid >> 5;
    const int lane = tid & 31;
    const int r16  = lane & 15;
    const int hi   = lane >> 4;

    // h0 = c0 = 0 (pad columns of h stay zero forever)
    for (int i = tid; i < BATCH * LDSROW; i += 1024) h_lds[i] = (bf16_t)0.f;
    for (int i = tid; i < BATCH * HID;    i += 1024) c_lds[i] = 0.f;
    __syncthreads();

    for (int t = 0; t < TSTEP; ++t) {
        // Prefetch next step's Xg block (300 KB) into the WGP while the
        // matrix pipe is busy; stage 2 of step t+1 will hit the cache.
        if (t + 1 < TSTEP) {
            const char* nx = (const char*)(Xg + (size_t)(t + 1) * BATCH * G4H);
            for (int p = tid; p < (BATCH * G4H * 4) / 256; p += 1024)
                __builtin_prefetch(nx + p * 256, 0, 0);
        }

        // ---- stage 1: 75 N-tiles over 32 waves; B fragment reused over 4 M-tiles
        for (int nt = wid; nt < 75; nt += 32) {
            const int n = nt * 16 + r16;
            const bf16_t* brow = WhhP + (size_t)n * HP;
            v8f acc[4] = {};
            #pragma unroll
            for (int kc = 0; kc < 10; ++kc) {
                Frag32B bfz;
                const u32x4* bp = (const u32x4*)(brow + kc * 32 + (hi << 4));
                bfz.q[0] = bp[0];
                bfz.q[1] = bp[1];
                #pragma unroll
                for (int mt = 0; mt < 4; ++mt) {
                    Frag32B af;
                    const bf16_t* ap =
                        h_lds + (size_t)(mt * 16 + r16) * LDSROW + kc * 32 + (hi << 4);
                    af.q[0] = *(const u32x4*)(ap);
                    af.q[1] = *(const u32x4*)(ap + 8);
                    acc[mt] = __builtin_amdgcn_wmma_f32_16x16x32_bf16(
                        false, af.v, false, bfz.v, (short)0, acc[mt], false, false);
                }
            }
            // D tile -> gate scratch in LDS (bf16): row m = mt*16+hi*8+r, col n
            #pragma unroll
            for (int mt = 0; mt < 4; ++mt) {
                const int mbase = mt * 16 + (hi << 3);
                #pragma unroll
                for (int r = 0; r < 8; ++r)
                    g_lds[(size_t)(mbase + r) * GROW + n] = (bf16_t)acc[mt][r];
            }
        }
        __syncthreads();

        // ---- stage 2: elementwise LSTM cell (i,f,g,o at same hidden index)
        const float* xrow = Xg + (size_t)t * BATCH * G4H;
        for (int e = tid; e < BATCH * HID; e += 1024) {
            const int b = e / HID;
            const int j = e - b * HID;
            const float*  xb = xrow  + (size_t)b * G4H;
            const bf16_t* gb = g_lds + (size_t)b * GROW;
            const float gi = xb[j]           + (float)gb[j];
            const float gf = xb[HID + j]     + (float)gb[HID + j];
            const float gg = xb[2 * HID + j] + (float)gb[2 * HID + j];
            const float go = xb[3 * HID + j] + (float)gb[3 * HID + j];
            const float i_ = sigf(gi);
            const float f_ = sigf(gf);
            const float g_ = tanhf(gg);
            const float o_ = sigf(go);
            const float c_ = f_ * c_lds[e] + i_ * g_;
            c_lds[e] = c_;
            h_lds[(size_t)b * LDSROW + j] = (bf16_t)(o_ * tanhf(c_));
        }
        __syncthreads();
    }

    // ---- final linear: out[b][k] = h_last[b] . W_lin[k] + b_lin[k]
    if (tid < 2 * BATCH) {
        const int b = tid >> 1;
        const int k = tid & 1;
        float s = b_lin[k];
        const float*  wl = W_lin + (size_t)k * HID;
        const bf16_t* hb = h_lds + (size_t)b * LDSROW;
        for (int j = 0; j < HID; ++j) s += (float)hb[j] * wl[j];
        out[b * 2 + k] = s;
    }
}

// ---------------------------------------------------------------------------
// Host launcher
// ---------------------------------------------------------------------------
extern "C" void kernel_launch(void* const* d_in, const int* in_sizes, int n_in,
                              void* d_out, int out_size, void* d_ws, size_t ws_size,
                              hipStream_t stream) {
    const int*   tokens = (const int*)  d_in[0];   // [256,64]
    const float* embed  = (const float*)d_in[1];   // [400000,300]
    const float* W_ih   = (const float*)d_in[2];   // [1200,300]
    const float* W_hh   = (const float*)d_in[3];   // [1200,300]
    const float* b_ih   = (const float*)d_in[4];   // [1200]
    const float* b_hh   = (const float*)d_in[5];   // [1200]
    const float* W_lin  = (const float*)d_in[6];   // [2,300]
    const float* b_lin  = (const float*)d_in[7];   // [2]
    float* out = (float*)d_out;                    // [1,64,2]

    // workspace layout (all offsets 256B-aligned)
    char* ws = (char*)d_ws;
    float*  Xg   = (float*)ws;                                   // 16384*1200*4 = 78,643,200 B
    size_t  o1   = (size_t)16384 * G4H * 4;
    bf16_t* WihP = (bf16_t*)(ws + o1);                           // 1200*320*2 = 768,000 B
    bf16_t* WhhP = (bf16_t*)(ws + o1 + 768000);                  // 768,000 B

    // 0) weight conversion (parallel, tiny)
    k_convert_w<<<(G4H * HP + 255) / 256, 256, 0, stream>>>(W_ih, W_hh, WihP, WhhP);

    // 1) parallel gate precompute: 1024 M-tiles * 15 N-groups = 15360 waves
    k_gates_x<<<15360 / 8, 256, 0, stream>>>(tokens, embed, WihP, b_ih, b_hh, Xg);

    // 2) serial recurrence: one persistent workgroup, 32 waves, big LDS
    const size_t lds_bytes = (size_t)BATCH * LDSROW * 2    // h     (bf16)  41,984 B
                           + (size_t)BATCH * HID * 4       // c     (f32)   76,800 B
                           + (size_t)BATCH * GROW * 2;     // gates (bf16) 155,648 B
    k_recurrence<<<1, 1024, lds_bytes, stream>>>(Xg, WhhP, W_lin, b_lin, out);
}
